// T5Attention_12970801234305
// MI455X (gfx1250) — matrix operand order repaired and checked by reading, not compiled, not run on hardware
//
#include <hip/hip_runtime.h>
#include <cmath>

// ---------- types ----------
typedef __attribute__((ext_vector_type(16))) __bf16 v16bf;
typedef __attribute__((ext_vector_type(8)))  float  v8f;

#define DEV static __device__ __forceinline__

// ---------- WMMA wrapper: D = A(16x32 bf16) x B(32x16 bf16) + C(f32) ----------
DEV v8f wmma_bf16(v16bf a, v16bf b, v8f c) {
  return __builtin_amdgcn_wmma_f32_16x16x32_bf16(false, a, false, b, (short)0, c,
                                                 false, false);
}

// A-fragment (16x32, 16-bit): lane holds row = lane&15.
// lanes 0-15: K = k0+{0..7}, k0+16+{0..7}; lanes 16-31: K = k0+8+{0..7}, k0+24+{0..7}
DEV v16bf load_a16(const __bf16* rowp, int k0, int half) {
  union { uint4 u[2]; v16bf v; } t;
  t.u[0] = *(const uint4*)(rowp + k0 + half * 8);
  t.u[1] = *(const uint4*)(rowp + k0 + 16 + half * 8);
  return t.v;
}

// B-fragment (32x16, 16-bit), data stored transposed (col-major = rows of Bt):
// lane holds column n = lane&15; lanes 0-15: K = k0+{0..7}, k0+16+{0..7}; lanes 16-31: K = k0+8+{0..7}, k0+24+{0..7}
DEV v16bf load_b16(const __bf16* colp, int k0, int half) {
  union { uint4 u[2]; v16bf v; } t;
  const __bf16* p = colp + k0 + half * 8;
  t.u[0] = *(const uint4*)(p);
  t.u[1] = *(const uint4*)(p + 16);
  return t.v;
}

// ---------- fp32 -> bf16 convert ----------
__global__ void f32_to_bf16_k(const float* __restrict__ in, __bf16* __restrict__ out, int n) {
  int i = blockIdx.x * blockDim.x + threadIdx.x;
  if (i < n) out[i] = (__bf16)in[i];
}

// ---------- transpose + convert: wt[n*dim+k] = (bf16) w[k*dim+n] ----------
__global__ void transpose_conv(const float* __restrict__ w, __bf16* __restrict__ wt, int dim) {
  __shared__ float tile[32][33];
  int tilesX = dim >> 5;
  int bx = blockIdx.x % tilesX;  // column-tile of w
  int by = blockIdx.x / tilesX;  // row-tile of w
  int x = bx * 32 + threadIdx.x;
  for (int i = threadIdx.y; i < 32; i += 8) {
    int y = by * 32 + i;
    tile[i][threadIdx.x] = w[(size_t)y * dim + x];
  }
  __syncthreads();
  int xo = by * 32 + threadIdx.x;  // k index (col of wt)
  for (int i = threadIdx.y; i < 32; i += 8) {
    int yo = bx * 32 + i;          // n index (row of wt)
    wt[(size_t)yo * dim + xo] = (__bf16)tile[threadIdx.x][i];
  }
}

// ---------- bf16 WMMA GEMM: C[M,N] = A[M,K] x Bt[N,K]^T ----------
// mode 0: Cout = float row-major [M,N]
// mode 1: Cout = bf16 [b, h, L, 64]   (Q/K head layout; row = b*2048+l, col = h*64+d)
// mode 2: Cout = bf16 [b, h, 64, L]   (V transposed)
__global__ void __launch_bounds__(128)
gemm_bf16_wmma(const __bf16* __restrict__ A, const __bf16* __restrict__ Bt,
               void* __restrict__ Cout, int M, int N, int K, int mode) {
  int lane = threadIdx.x & 31;
  int half = lane >> 4;
  int r15  = lane & 15;
  int gw = blockIdx.x * 4 + (threadIdx.x >> 5);
  int tilesN = N >> 5;
  int tm = gw / tilesN, tn = gw % tilesN;
  if (tm * 32 >= M) return;
  int m0 = tm * 32, n0 = tn * 32;

  const __bf16* arow0 = A  + (size_t)(m0 + r15) * K;
  const __bf16* arow1 = arow0 + (size_t)16 * K;
  const __bf16* brow0 = Bt + (size_t)(n0 + r15) * K;
  const __bf16* brow1 = brow0 + (size_t)16 * K;

  v8f acc[2][2] = {};
  for (int k0 = 0; k0 < K; k0 += 32) {
    v16bf a0 = load_a16(arow0, k0, half);
    v16bf a1 = load_a16(arow1, k0, half);
    v16bf b0 = load_b16(brow0, k0, half);
    v16bf b1 = load_b16(brow1, k0, half);
    acc[0][0] = wmma_bf16(a0, b0, acc[0][0]);
    acc[0][1] = wmma_bf16(a0, b1, acc[0][1]);
    acc[1][0] = wmma_bf16(a1, b0, acc[1][0]);
    acc[1][1] = wmma_bf16(a1, b1, acc[1][1]);
  }

#pragma unroll
  for (int i = 0; i < 2; ++i)
#pragma unroll
    for (int j = 0; j < 2; ++j)
#pragma unroll
      for (int r = 0; r < 8; ++r) {
        int grow = m0 + i * 16 + r + half * 8;   // C layout: row = r + 8*(lane/16)
        int gcol = n0 + j * 16 + r15;            // col = lane&15
        float v = acc[i][j][r];
        if (mode == 0) {
          ((float*)Cout)[(size_t)grow * N + gcol] = v;
        } else {
          int bb = grow >> 11, lq = grow & 2047;
          int hh = gcol >> 6,  d  = gcol & 63;
          if (mode == 1)
            ((__bf16*)Cout)[((size_t)(bb * 16 + hh) * 2048 + lq) * 64 + d] = (__bf16)v;
          else
            ((__bf16*)Cout)[((size_t)(bb * 16 + hh) * 64 + d) * 2048 + lq] = (__bf16)v;
        }
      }
}

// ---------- flash attention: one 64-row query block per 128-thread block ----------
// blockIdx.x = qb*64 + h*4 + b  (batch fastest -> 4 batches share bias tile in L2)
__global__ void __launch_bounds__(128)
flash_t5(const __bf16* __restrict__ Q, const __bf16* __restrict__ Kh,
         const __bf16* __restrict__ Vt, const float* __restrict__ PB,
         const int* __restrict__ MSK, __bf16* __restrict__ Oout) {
  __shared__ __align__(16) __bf16 lds_p[4][16][32];

  int lane = threadIdx.x & 31, half = lane >> 4, r15 = lane & 15;
  int wave = threadIdx.x >> 5;
  int b  = blockIdx.x & 3;
  int h  = (blockIdx.x >> 2) & 15;
  int qb = blockIdx.x >> 6;
  int q0 = qb * 64 + wave * 16;

  size_t headoff = (size_t)(b * 16 + h) * 2048 * 64;
  const __bf16* qh = Q  + headoff;
  const __bf16* kh = Kh + headoff;
  const __bf16* vh = Vt + headoff;                 // [64][2048] per head
  const float*  pb = PB + (size_t)h * 2048 * 2048; // broadcast over batch
  const int*    mrow = MSK + b * 2048;

  const __bf16* qrowp = qh + (size_t)(q0 + r15) * 64;
  v16bf qa0 = load_a16(qrowp, 0, half);
  v16bf qa1 = load_a16(qrowp, 32, half);

  v8f accO[4] = {};
  float m[8], l[8];
#pragma unroll
  for (int r = 0; r < 8; ++r) { m[r] = -__builtin_inff(); l[r] = 0.f; }

  for (int j0 = 0; j0 < 2048; j0 += 32) {
    // ---- S = Q x K^T (+bias, +mask), two 16-wide column subtiles ----
    v8f s[2];
#pragma unroll
    for (int nt = 0; nt < 2; ++nt) {
      const __bf16* kp = kh + (size_t)(j0 + nt * 16 + r15) * 64;
      v16bf kf0 = load_b16(kp, 0, half);
      v16bf kf1 = load_b16(kp, 32, half);
      v8f a = {};
      a = wmma_bf16(qa0, kf0, a);
      a = wmma_bf16(qa1, kf1, a);
      int col = j0 + nt * 16 + r15;
      bool ok = (mrow[col] != 0);
#pragma unroll
      for (int r = 0; r < 8; ++r) {
        int row = q0 + r + half * 8;
        s[nt][r] = ok ? (a[r] + pb[(size_t)row * 2048 + col]) : -3.402823466e38f;
      }
    }
    // ---- online softmax (row stats across 16 lanes of each half-wave) ----
#pragma unroll
    for (int r = 0; r < 8; ++r) {
      float mx = fmaxf(s[0][r], s[1][r]);
      mx = fmaxf(mx, __shfl_xor(mx, 1));
      mx = fmaxf(mx, __shfl_xor(mx, 2));
      mx = fmaxf(mx, __shfl_xor(mx, 4));
      mx = fmaxf(mx, __shfl_xor(mx, 8));
      float mn = fmaxf(m[r], mx);
      float scale = __expf(m[r] - mn);
      m[r] = mn;
      float p0 = __expf(s[0][r] - mn);
      float p1 = __expf(s[1][r] - mn);
      float rs = p0 + p1;
      rs += __shfl_xor(rs, 1);
      rs += __shfl_xor(rs, 2);
      rs += __shfl_xor(rs, 4);
      rs += __shfl_xor(rs, 8);
      l[r] = l[r] * scale + rs;
#pragma unroll
      for (int nt = 0; nt < 4; ++nt) accO[nt][r] *= scale;
      // stage P (C layout -> LDS, bf16) for the A-operand of P x V
      lds_p[wave][r + half * 8][r15]      = (__bf16)p0;
      lds_p[wave][r + half * 8][r15 + 16] = (__bf16)p1;
    }
    __syncthreads();
    v16bf pf = load_a16(&lds_p[wave][r15][0], 0, half);
#pragma unroll
    for (int nt = 0; nt < 4; ++nt) {
      v16bf vf = load_b16(vh + (size_t)(nt * 16 + r15) * 2048, j0, half);
      accO[nt] = wmma_bf16(pf, vf, accO[nt]);
    }
    __syncthreads();
  }

  // ---- normalize and write attn output as bf16 [b, l, h*64+d] ----
#pragma unroll
  for (int nt = 0; nt < 4; ++nt)
#pragma unroll
    for (int r = 0; r < 8; ++r) {
      int row = q0 + r + half * 8;
      int d = h * 64 + nt * 16 + r15;
      Oout[((size_t)b * 2048 + row) * 1024 + d] = (__bf16)(accO[nt][r] / l[r]);
    }
}

// ---------- host launch ----------
extern "C" void kernel_launch(void* const* d_in, const int* in_sizes, int n_in,
                              void* d_out, int out_size, void* d_ws, size_t ws_size,
                              hipStream_t stream) {
  const float* x  = (const float*)d_in[0];
  const float* wq = (const float*)d_in[1];
  const float* wk = (const float*)d_in[2];
  const float* wv = (const float*)d_in[3];
  const float* wo = (const float*)d_in[4];
  const float* pb = (const float*)d_in[5];
  const int*   mk = (const int*)d_in[6];
  float* out = (float*)d_out;

  const int M = 4 * 2048;          // 8192 rows
  const int DIMD = 1024;
  const size_t nX = (size_t)M * DIMD;        // 8,388,608
  const size_t nW = (size_t)DIMD * DIMD;     // 1,048,576

  char* p = (char*)d_ws;
  __bf16* xb  = (__bf16*)p; p += nX * 2;
  __bf16* wqT = (__bf16*)p; p += nW * 2;
  __bf16* wkT = (__bf16*)p; p += nW * 2;
  __bf16* wvT = (__bf16*)p; p += nW * 2;
  __bf16* woT = (__bf16*)p; p += nW * 2;
  __bf16* qws = (__bf16*)p; p += nX * 2;
  __bf16* kws = (__bf16*)p; p += nX * 2;
  __bf16* vtw = (__bf16*)p; p += nX * 2;
  __bf16* aws = (__bf16*)p; p += nX * 2;

  // 1) convert x to bf16
  f32_to_bf16_k<<<(int)(nX / 256), 256, 0, stream>>>(x, xb, (int)nX);
  // 2) transpose+convert weights
  dim3 tb(32, 8);
  int tgrid = (DIMD / 32) * (DIMD / 32);
  transpose_conv<<<tgrid, tb, 0, stream>>>(wq, wqT, DIMD);
  transpose_conv<<<tgrid, tb, 0, stream>>>(wk, wkT, DIMD);
  transpose_conv<<<tgrid, tb, 0, stream>>>(wv, wvT, DIMD);
  transpose_conv<<<tgrid, tb, 0, stream>>>(wo, woT, DIMD);
  // 3) QKV projections: waves = (M/32)*(N/32) = 8192 -> 2048 blocks of 4 waves
  int gblocks = ((M / 32) * (DIMD / 32)) / 4;
  gemm_bf16_wmma<<<gblocks, 128, 0, stream>>>(xb, wqT, qws, M, DIMD, DIMD, 1);
  gemm_bf16_wmma<<<gblocks, 128, 0, stream>>>(xb, wkT, kws, M, DIMD, DIMD, 1);
  gemm_bf16_wmma<<<gblocks, 128, 0, stream>>>(xb, wvT, vtw, M, DIMD, DIMD, 2);
  // 4) flash attention: 32 qblocks * 16 heads * 4 batches = 2048 blocks
  flash_t5<<<2048, 128, 0, stream>>>(qws, kws, vtw, pb, mk, aws);
  // 5) output projection -> fp32 d_out
  gemm_bf16_wmma<<<gblocks, 128, 0, stream>>>(aws, woT, out, M, DIMD, DIMD, 0);
}
